// Q_NsNet2_npy_83365315215849
// MI455X (gfx1250) — 8-bit matrix operand order repaired and checked by reading, compiled, not run on hardware
//
#include <hip/hip_runtime.h>
#include <hip/hip_bf16.h>

typedef __attribute__((ext_vector_type(8))) int v8i;

#define NB   4096
#define BM   64
#define BN   128
#define BK   64

__device__ __forceinline__ void async_load_b128(void* lds, const void* gaddr)
{
    unsigned l = (unsigned)(unsigned long long)(uintptr_t)lds;
    asm volatile("global_load_async_to_lds_b128 %0, %1, off"
                 :: "v"(l), "v"(gaddr) : "memory");
}

__device__ __forceinline__ void wait_async0()
{
#if __has_builtin(__builtin_amdgcn_s_wait_asynccnt)
    __builtin_amdgcn_s_wait_asynccnt(0);
#else
    asm volatile("s_wait_asynccnt 0x0" ::: "memory");
#endif
}

__device__ __forceinline__ float fast_sigmoid(float x)
{
    return __builtin_amdgcn_rcpf(1.0f + __expf(-x));
}

__device__ __forceinline__ float amax_to_scale(unsigned bits)
{
    return __uint_as_float(bits) * (1.0f / 127.0f) + 1e-12f;
}

__global__ __launch_bounds__(256) void absmax_kernel(
    const float* __restrict__ x, int n, unsigned* __restrict__ amax)
{
    __shared__ float red[256];
    float m = 0.0f;
    for (int i = blockIdx.x * blockDim.x + threadIdx.x; i < n;
         i += gridDim.x * blockDim.x)
        m = fmaxf(m, fabsf(x[i]));
    red[threadIdx.x] = m;
    __syncthreads();
    for (int s = 128; s > 0; s >>= 1) {
        if (threadIdx.x < s)
            red[threadIdx.x] = fmaxf(red[threadIdx.x], red[threadIdx.x + s]);
        __syncthreads();
    }
    if (threadIdx.x == 0) atomicMax(amax, __float_as_uint(red[0]));
}

__global__ __launch_bounds__(256) void quant_w_kernel(
    const float* __restrict__ W, int M, int K, int Kpad,
    const unsigned* __restrict__ amax, signed char* __restrict__ out)
{
    int idx = blockIdx.x * blockDim.x + threadIdx.x;
    int total = M * Kpad;
    if (idx >= total) return;
    int m = idx / Kpad, k = idx - m * Kpad;
    float inv = 1.0f / amax_to_scale(*amax);
    float v = (k < K) ? W[(size_t)m * K + k] : 0.0f;
    float q = rintf(v * inv);
    q = fminf(127.0f, fmaxf(-127.0f, q));
    out[idx] = (signed char)q;
}

__global__ __launch_bounds__(256) void quant_at_kernel(
    const float* __restrict__ X, int K, int Kpad,
    const unsigned* __restrict__ amax, signed char* __restrict__ outT)
{
    int idx = blockIdx.x * blockDim.x + threadIdx.x;
    int total = NB * Kpad;
    if (idx >= total) return;
    int n = idx / Kpad, k = idx - n * Kpad;
    float inv = 1.0f / amax_to_scale(*amax);
    float v = (k < K) ? X[(size_t)k * NB + n] : 0.0f;
    float q = rintf(v * inv);
    q = fminf(127.0f, fmaxf(-127.0f, q));
    outT[idx] = (signed char)q;
}

__global__ __launch_bounds__(256, 2) void gemm_iu8_kernel(
    const signed char* __restrict__ A,
    const signed char* __restrict__ Bt,
    const float* __restrict__ bias,
    const unsigned* __restrict__ amaxA, const unsigned* __restrict__ amaxB,
    float* __restrict__ C,
    int M, int Kpad, int act)
{
    __shared__ __align__(16) signed char As[2][BM * BK];
    __shared__ __align__(16) signed char Bs[2][BN * BK];

    const int tid   = threadIdx.x;
    const int lane  = tid & 31;
    const int wave  = tid >> 5;
    const int waveM = wave & 1;
    const int waveN = wave >> 1;
    const int mBase = blockIdx.y * BM;
    const int nBase = blockIdx.x * BN;

    const int rA = tid >> 2, oA = (tid & 3) << 4;
    int gmA = mBase + rA; if (gmA >= M) gmA = M - 1;
    const signed char* srcA = A + (size_t)gmA * Kpad + oA;

    auto issue_tile = [&](int k0, int buf) {
        async_load_b128(As[buf] + rA * BK + oA, srcA + k0);
        #pragma unroll
        for (int i = 0; i < 2; ++i) {
            int chunk = tid + (i << 8);
            int r = chunk >> 2, o = (chunk & 3) << 4;
            async_load_b128(Bs[buf] + r * BK + o,
                            Bt + (size_t)(nBase + r) * Kpad + k0 + o);
        }
    };

    const v8i vzero = {0, 0, 0, 0, 0, 0, 0, 0};
    v8i acc[2][2];
    acc[0][0] = vzero; acc[0][1] = vzero; acc[1][0] = vzero; acc[1][1] = vzero;

    issue_tile(0, 0);
    int buf = 0;
    for (int k0 = 0; k0 < Kpad; k0 += BK) {
        wait_async0();
        __syncthreads();
        if (k0 + BK < Kpad)
            issue_tile(k0 + BK, buf ^ 1);

        v8i afrag[2], bfrag[2];
        const int mrow = lane & 15;
        const int khalfA = (lane >> 4) << 3;
        #pragma unroll
        for (int t = 0; t < 2; ++t) {
            const signed char* base = As[buf] + (waveM * 32 + t * 16 + mrow) * BK;
            v8i a = vzero;
            #pragma unroll
            for (int v = 0; v < 8; ++v) {
                int k = ((v >> 2) << 5) + (khalfA << 1) + ((v & 3) << 2);
                a[v] = *reinterpret_cast<const int*>(base + k);
            }
            afrag[t] = a;
        }
        const int khalfB = (lane >> 4) << 4;
        #pragma unroll
        for (int t = 0; t < 2; ++t) {
            const signed char* base = Bs[buf] + (waveN * 32 + t * 16 + (lane & 15)) * BK;
            v8i b = vzero;
            #pragma unroll
            for (int v = 0; v < 8; ++v) {
                int k = ((v >> 2) << 5) + khalfB + ((v & 3) << 2);
                b[v] = *reinterpret_cast<const int*>(base + k);
            }
            bfrag[t] = b;
        }

        #pragma unroll
        for (int tm = 0; tm < 2; ++tm)
            #pragma unroll
            for (int tn = 0; tn < 2; ++tn)
                acc[tm][tn] = __builtin_amdgcn_wmma_i32_16x16x64_iu8(
                    true, afrag[tm], true, bfrag[tn], acc[tm][tn], false, false);

        buf ^= 1;
    }

    const float scale = amax_to_scale(*amaxA) * amax_to_scale(*amaxB);
    const int rowHi = (lane >> 4) << 3;
    const int coln  = lane & 15;
    #pragma unroll
    for (int tm = 0; tm < 2; ++tm) {
        #pragma unroll
        for (int tn = 0; tn < 2; ++tn) {
            #pragma unroll
            for (int r = 0; r < 8; ++r) {
                int m = mBase + waveM * 32 + tm * 16 + rowHi + r;
                if (m < M) {
                    int n = nBase + waveN * 32 + tn * 16 + coln;
                    float f = (float)acc[tm][tn][r] * scale + bias[m];
                    if (act == 1)      f = fmaxf(f, 0.0f);
                    else if (act == 2) f = fast_sigmoid(f);
                    C[(size_t)m * NB + n] = f;
                }
            }
        }
    }
}

__global__ __launch_bounds__(256) void gru_combine_kernel(
    const float* __restrict__ a, const float* __restrict__ b,
    const float* __restrict__ c, const float* __restrict__ d,
    const float* __restrict__ e, const float* __restrict__ f,
    const float* __restrict__ hprev, float* __restrict__ hout, int n)
{
    int i = blockIdx.x * blockDim.x + threadIdx.x;
    if (i >= n) return;
    float r  = fast_sigmoid(a[i] + b[i]);
    float z  = fast_sigmoid(c[i] + d[i]);
    float nn = tanhf(e[i] + r * f[i]);
    hout[i]  = (1.0f - z) * nn + z * hprev[i];
}

extern "C" void kernel_launch(void* const* d_in, const int* in_sizes, int n_in,
                              void* d_out, int out_size, void* d_ws, size_t ws_size,
                              hipStream_t stream)
{
    (void)in_sizes; (void)n_in; (void)out_size; (void)ws_size;
    const int F = 257, H = 400, D = 600;
    const int Fp = 320, Hp = 448, Dp = 640;

    const float* X  = (const float*)d_in[0];
    const float* h1 = (const float*)d_in[1];
    const float* h2 = (const float*)d_in[2];

    const int widx[16] = {3, 5, 6, 9, 10, 13, 14, 17, 18, 21, 22, 25, 26, 29, 31, 33};
    int wM[16], wK[16], wKp[16];
    for (int i = 0; i < 16; ++i) { wM[i] = H; wK[i] = H; wKp[i] = Hp; }
    wM[0] = H; wK[0] = F; wKp[0] = Fp;
    wM[13] = D; wK[13] = H; wKp[13] = Hp;
    wM[14] = D; wK[14] = D; wKp[14] = Dp;
    wM[15] = F; wK[15] = D; wKp[15] = Dp;

    char* ws = (char*)d_ws;
    size_t cur = 0;
    auto alloc = [&](size_t bytes) -> char* {
        char* p = ws + cur;
        cur += (bytes + 255) & ~(size_t)255;
        return p;
    };

    enum { AX = 16, AFC1 = 17, AH1 = 18, AG1 = 19, AH2 = 20, AG2 = 21, AA2 = 22, AA3 = 23 };
    unsigned* amax = (unsigned*)alloc(32 * sizeof(unsigned));

    signed char* Wq[16];
    for (int i = 0; i < 16; ++i)
        Wq[i] = (signed char*)alloc((size_t)wM[i] * wKp[i]);

    signed char* xqT   = (signed char*)alloc((size_t)NB * Fp);
    signed char* xinqT = (signed char*)alloc((size_t)NB * Hp);
    signed char* hqT   = (signed char*)alloc((size_t)NB * Hp);
    signed char* dqT   = (signed char*)alloc((size_t)NB * Dp);

    float* fc1buf  = (float*)alloc((size_t)H * NB * 4);
    float* gate[6];
    for (int i = 0; i < 6; ++i) gate[i] = (float*)alloc((size_t)H * NB * 4);
    float* dBuf = (float*)alloc((size_t)D * NB * 4);
    float* outF = (float*)d_out;

    hipMemsetAsync(amax, 0, 32 * sizeof(unsigned), stream);

    auto absmax = [&](const float* x, int n, int slot) {
        int blocks = (n + 256 * 8 - 1) / (256 * 8);
        if (blocks < 1) blocks = 1;
        if (blocks > 1024) blocks = 1024;
        absmax_kernel<<<blocks, 256, 0, stream>>>(x, n, amax + slot);
    };
    auto quantW = [&](int i) {
        int total = wM[i] * wKp[i];
        quant_w_kernel<<<(total + 255) / 256, 256, 0, stream>>>(
            (const float*)d_in[widx[i]], wM[i], wK[i], wKp[i], amax + i, Wq[i]);
    };
    auto quantAT = [&](const float* x, int K, int Kp, int slot, signed char* out) {
        int total = NB * Kp;
        quant_at_kernel<<<(total + 255) / 256, 256, 0, stream>>>(
            x, K, Kp, amax + slot, out);
    };
    auto gemm = [&](int wi, const signed char* bt, const float* bias, int slotB,
                    float* Cout, int act) {
        dim3 grid(NB / BN, (wM[wi] + BM - 1) / BM);
        gemm_iu8_kernel<<<grid, 256, 0, stream>>>(
            Wq[wi], bt, bias, amax + wi, amax + slotB, Cout, wM[wi], wKp[wi], act);
    };

    for (int i = 0; i < 16; ++i) absmax((const float*)d_in[widx[i]], wM[i] * wK[i], i);
    absmax(X, F * NB, AX);
    for (int i = 0; i < 16; ++i) quantW(i);
    quantAT(X, F, Fp, AX, xqT);

    gemm(0, xqT, (const float*)d_in[4], AX, fc1buf, 0);

    absmax(fc1buf, H * NB, AFC1);
    quantAT(fc1buf, H, Hp, AFC1, xinqT);
    absmax(h1, H * NB, AH1);
    quantAT(h1, H, Hp, AH1, hqT);
    gemm(1, xinqT, (const float*)d_in[7],  AFC1, gate[0], 0);
    gemm(2, hqT,   (const float*)d_in[8],  AH1,  gate[1], 0);
    gemm(3, xinqT, (const float*)d_in[11], AFC1, gate[2], 0);
    gemm(4, hqT,   (const float*)d_in[12], AH1,  gate[3], 0);
    gemm(5, xinqT, (const float*)d_in[15], AFC1, gate[4], 0);
    gemm(6, hqT,   (const float*)d_in[16], AH1,  gate[5], 0);
    {
        int n = H * NB;
        gru_combine_kernel<<<(n + 255) / 256, 256, 0, stream>>>(
            gate[0], gate[1], gate[2], gate[3], gate[4], gate[5], h1, fc1buf, n);
    }

    absmax(fc1buf, H * NB, AG1);
    quantAT(fc1buf, H, Hp, AG1, xinqT);
    absmax(h2, H * NB, AH2);
    quantAT(h2, H, Hp, AH2, hqT);
    gemm(7,  xinqT, (const float*)d_in[19], AG1, gate[0], 0);
    gemm(8,  hqT,   (const float*)d_in[20], AH2, gate[1], 0);
    gemm(9,  xinqT, (const float*)d_in[23], AG1, gate[2], 0);
    gemm(10, hqT,   (const float*)d_in[24], AH2, gate[3], 0);
    gemm(11, xinqT, (const float*)d_in[27], AG1, gate[4], 0);
    gemm(12, hqT,   (const float*)d_in[28], AH2, gate[5], 0);
    {
        int n = H * NB;
        gru_combine_kernel<<<(n + 255) / 256, 256, 0, stream>>>(
            gate[0], gate[1], gate[2], gate[3], gate[4], gate[5], h2, fc1buf, n);
    }

    absmax(fc1buf, H * NB, AG2);
    quantAT(fc1buf, H, Hp, AG2, xinqT);
    gemm(13, xinqT, (const float*)d_in[30], AG2, dBuf, 1);

    absmax(dBuf, D * NB, AA2);
    quantAT(dBuf, D, Dp, AA2, dqT);
    gemm(14, dqT, (const float*)d_in[32], AA2, dBuf, 1);

    absmax(dBuf, D * NB, AA3);
    quantAT(dBuf, D, Dp, AA3, dqT);
    gemm(15, dqT, (const float*)d_in[34], AA3, outF, 2);
}
